// EnhancedStateEncoder_6897717477674
// MI455X (gfx1250) — compile-verified, matrix operand order checked by reading, not run on hardware
//
#include <hip/hip_runtime.h>
#include <hip/hip_bf16.h>
#include <math.h>
#include <stdint.h>

typedef __attribute__((ext_vector_type(16))) _Float16     v16h;
typedef __attribute__((ext_vector_type(8)))  float        v8f;
typedef __attribute__((ext_vector_type(4)))  unsigned int u32x4;
typedef __attribute__((ext_vector_type(8)))  int          i32x8;
typedef __attribute__((ext_vector_type(4)))  int          i32x4;

#define EMB_DIM  256
#define HEADS    8
#define HEAD_DIM 32
#define N_LAYERS 6
#define SEQ      1024
#define BATCH    16
#define MLP_H    1024
#define ROWS     (BATCH * SEQ)
#define LN_EPS   1e-5f

#if defined(__has_builtin)
#  if __has_builtin(__builtin_amdgcn_tensor_load_to_lds)
#    define HAVE_TDM 1
#  endif
#  if __has_builtin(__builtin_amdgcn_s_wait_tensorcnt)
#    define HAVE_TENSORCNT 1
#  endif
#endif

// ---------------------------------------------------------------------------
// WMMA helpers (CDNA5: V_WMMA_F32_16X16X32_F16, wave32)
// ---------------------------------------------------------------------------
static __device__ __forceinline__ v8f wmma16(v16h a, v16h b, v8f c) {
  return __builtin_amdgcn_wmma_f32_16x16x32_f16(
      /*neg_a=*/false, a, /*neg_b=*/false, b,
      /*c_mod=*/(short)0, c, /*reuse_a=*/false, /*reuse_b=*/false);
}

// Load a 16x32 f16 tile (row-major, leading dim `ld` elements) into the
// per-lane A/B operand layout (16-bit A-matrix layout, ISA 7.12.2):
//   lanes 0-15 : row m=lane,    K = 0..7 (v0-3) and 16..23 (v4-7)
//   lanes16-31 : row m=lane-16, K = 8..15(v0-3) and 24..31 (v4-7)
// For the B operand we pass the *transposed* matrix (lane = column).
static __device__ __forceinline__ v16h load_tile(const _Float16* __restrict__ p,
                                                 int ld, int lane) {
  int m = lane & 15, half = lane >> 4;
  const _Float16* r0 = p + (size_t)m * ld + 8 * half;
  v16h a;
#pragma unroll
  for (int e = 0; e < 8; ++e) a[e] = r0[e];
#pragma unroll
  for (int e = 0; e < 8; ++e) a[8 + e] = r0[16 + e];
  return a;
}

// ---------------------------------------------------------------------------
// Tensor Data Mover: DMA a 16x32 f16 tile (row pitch `ld` elems) into LDS.
// D# per ISA 08_async_tensor: group0 = {count, lds_addr, global_addr, type=2},
// group1 = {data_size=2B, tensor_dim0=ld, tensor_dim1=ROWS, tile 32x16,
// tensor_dim0_stride=ld}. Issued by one wave; EXEC is ignored by TDM.
// ---------------------------------------------------------------------------
static __device__ __forceinline__ void tdm_load_tile16x32(
    _Float16* lds_dst, const _Float16* gsrc, unsigned ld, int lane) {
#if defined(HAVE_TDM)
  unsigned long long ga = (unsigned long long)(uintptr_t)gsrc;
  unsigned lds_off = (unsigned)(uintptr_t)lds_dst;  // LDS offset = addr[31:0]
  u32x4 g0;
  g0[0] = 1u;                                           // count=1, user D#
  g0[1] = lds_off;                                      // lds_addr (bytes)
  g0[2] = (unsigned)(ga & 0xFFFFFFFFu);                 // global_addr[31:0]
  g0[3] = (unsigned)((ga >> 32) & 0x01FFFFFFu) | 0x80000000u;  // [56:32]|type=2
  i32x8 g1;
  g1[0] = 1 << 16;                                      // data_size = 2 bytes
  g1[1] = (int)((ld & 0xFFFFu) << 16);                  // tensor_dim0[15:0]
  g1[2] = (int)((ld >> 16) | ((ROWS & 0xFFFFu) << 16)); // dim0 hi | dim1 lo
  g1[3] = (int)((ROWS >> 16) | (32u << 16));            // dim1 hi | tile_dim0=32
  g1[4] = 16;                                           // tile_dim1=16
  g1[5] = (int)ld;                                      // dim0_stride[31:0]
  g1[6] = 0;
  g1[7] = 0;
  i32x4 z4 = {};
#  if __clang_major__ >= 23
  i32x8 z8 = {};
  __builtin_amdgcn_tensor_load_to_lds(g0, g1, z4, z4, z8, 0);
#  else
  __builtin_amdgcn_tensor_load_to_lds(g0, g1, z4, z4, 0);
#  endif
#  if defined(HAVE_TENSORCNT)
  __builtin_amdgcn_s_wait_tensorcnt(0);
#  endif
#else
  // Fallback: plain cooperative copy by the issuing wave.
  for (int i = lane; i < 16 * 32; i += 32)
    lds_dst[i] = gsrc[(size_t)(i >> 5) * ld + (i & 31)];
#endif
}

// ---------------------------------------------------------------------------
// Embedding + sinusoidal PE mix
// ---------------------------------------------------------------------------
__global__ void embed_kernel(const int* __restrict__ tokens,
                             const int* __restrict__ pos_ids,
                             const float* __restrict__ emb,
                             const float* __restrict__ iw,
                             const float* __restrict__ pw,
                             float* __restrict__ x) {
  int idx = blockIdx.x * blockDim.x + threadIdx.x;
  if (idx >= BATCH * SEQ * EMB_DIM) return;
  int d = idx & (EMB_DIM - 1);
  int s = (idx >> 8) & (SEQ - 1);
  int b = idx >> 18;
  int tok = tokens[b * SEQ + s];
  float pos = (float)pos_ids[s];
  int i = (d < 128) ? d : d - 128;
  float freq = __powf(10000.0f, -(float)i * (1.0f / 128.0f));
  float ang = pos * freq;
  float pe = (d < 128) ? __sinf(ang) : __cosf(ang);
  x[idx] = iw[0] * emb[tok * EMB_DIM + d] + pw[0] * pe;
}

// ---------------------------------------------------------------------------
// LayerNorm: one wave per 256-elem row -> f16 row-major (xn) and, optionally,
// a head-transposed f16 copy xnT[B][H][HEAD_DIM][SEQ] used as the V^T operand.
// ---------------------------------------------------------------------------
__global__ void ln_f16_kernel(const float* __restrict__ x,
                              const float* __restrict__ w,
                              const float* __restrict__ b,
                              _Float16* __restrict__ out,
                              _Float16* __restrict__ outT,
                              int writeT) {
  int lane = threadIdx.x & 31;
  int wave = threadIdx.x >> 5;
  int row = blockIdx.x * 8 + wave;
  const float* xr = x + (size_t)row * EMB_DIM + lane * 8;
  float v[8];
  float s = 0.f;
#pragma unroll
  for (int i = 0; i < 8; ++i) { v[i] = xr[i]; s += v[i]; }
#pragma unroll
  for (int off = 1; off < 32; off <<= 1) s += __shfl_xor(s, off, 32);
  float mu = s * (1.0f / EMB_DIM);
  float q = 0.f;
#pragma unroll
  for (int i = 0; i < 8; ++i) { float d = v[i] - mu; q += d * d; }
#pragma unroll
  for (int off = 1; off < 32; off <<= 1) q += __shfl_xor(q, off, 32);
  float inv = rsqrtf(q * (1.0f / EMB_DIM) + LN_EPS);
  _Float16* orow = out + (size_t)row * EMB_DIM + lane * 8;
  int bidx = row >> 10, sidx = row & (SEQ - 1);
#pragma unroll
  for (int i = 0; i < 8; ++i) {
    int d = lane * 8 + i;
    _Float16 hv = (_Float16)((v[i] - mu) * inv * w[d] + b[d]);
    orow[i] = hv;
    if (writeT) {
      int head = d >> 5, dim = d & 31;
      outT[(((size_t)bidx * HEADS + head) * HEAD_DIM + dim) * SEQ + sidx] = hv;
    }
  }
}

__global__ void lnf_kernel(const float* __restrict__ x,
                           const float* __restrict__ w,
                           const float* __restrict__ b,
                           float* __restrict__ out) {
  int lane = threadIdx.x & 31;
  int wave = threadIdx.x >> 5;
  int row = blockIdx.x * 8 + wave;
  const float* xr = x + (size_t)row * EMB_DIM + lane * 8;
  float v[8];
  float s = 0.f;
#pragma unroll
  for (int i = 0; i < 8; ++i) { v[i] = xr[i]; s += v[i]; }
#pragma unroll
  for (int off = 1; off < 32; off <<= 1) s += __shfl_xor(s, off, 32);
  float mu = s * (1.0f / EMB_DIM);
  float q = 0.f;
#pragma unroll
  for (int i = 0; i < 8; ++i) { float d = v[i] - mu; q += d * d; }
#pragma unroll
  for (int off = 1; off < 32; off <<= 1) q += __shfl_xor(q, off, 32);
  float inv = rsqrtf(q * (1.0f / EMB_DIM) + LN_EPS);
  float* orow = out + (size_t)row * EMB_DIM + lane * 8;
#pragma unroll
  for (int i = 0; i < 8; ++i)
    orow[i] = (v[i] - mu) * inv * w[lane * 8 + i] + b[lane * 8 + i];
}

// ---------------------------------------------------------------------------
// Flash-style attention with 2D-ALiBi bias.
// One wave per (batch, head, 16 query rows). q=k from xn (row-major),
// v from xnT (dim-major -> contiguous B-operand loads).
// ---------------------------------------------------------------------------
__global__ void attn_kernel(const _Float16* __restrict__ xn,
                            const _Float16* __restrict__ xnT,
                            float* __restrict__ x) {
  __shared__ _Float16 P[16 * 32];  // probability tile: C-layout -> A-layout
  int lane = threadIdx.x;
  int qbase = blockIdx.x * 16;
  int head = blockIdx.y;
  int b = blockIdx.z;
  size_t bb = (size_t)b * SEQ * EMB_DIM;
  size_t vb_base = ((size_t)b * HEADS + head) * HEAD_DIM * SEQ;  // xnT
  int hb = head * HEAD_DIM;
  int half = lane >> 4, col = lane & 15;

  const float scale = 0.17677669529663687f;  // 1/sqrt(32)
  float slope_r = -exp2f(-(0.5f + (float)head));
  float slope_l = -exp2f(-(1.0f + (float)head));

  v16h qa = load_tile(xn + bb + (size_t)qbase * EMB_DIM + hb, EMB_DIM, lane);

  float m[8], l[8];
  int qx[8], qy[8];
  v8f o0 = {}, o1 = {};
#pragma unroll
  for (int r = 0; r < 8; ++r) {
    m[r] = -1e30f; l[r] = 0.f;
    int q = qbase + r + 8 * half;
    qx[r] = q >> 5; qy[r] = q & 31;
  }

  for (int t = 0; t < SEQ / 32; ++t) {
    int kbase = t * 32;
    if (t + 1 < SEQ / 32)  // global_prefetch_b8 the next key chunk
      __builtin_prefetch(xn + bb + (size_t)(kbase + 32) * EMB_DIM + hb, 0, 1);
    // --- scores for 32 keys: two 16x16 WMMA tiles -------------------------
    v16h kb0 = load_tile(xn + bb + (size_t)kbase * EMB_DIM + hb, EMB_DIM, lane);
    v16h kb1 = load_tile(xn + bb + (size_t)(kbase + 16) * EMB_DIM + hb, EMB_DIM, lane);
    v8f zero = {};
    v8f c0 = wmma16(qa, kb0, zero);
    v8f c1 = wmma16(qa, kb1, zero);

#pragma unroll
    for (int r = 0; r < 8; ++r) {
      int q = qbase + r + 8 * half;
      int k0 = kbase + col, k1 = kbase + 16 + col;
      float d0 = fabsf((float)qx[r] - (float)(k0 >> 5)) +
                 fabsf((float)qy[r] - (float)(k0 & 31));
      float d1 = fabsf((float)qx[r] - (float)(k1 >> 5)) +
                 fabsf((float)qy[r] - (float)(k1 & 31));
      float s0 = c0[r] * scale + ((k0 >= q) ? slope_r : slope_l) * d0;
      float s1 = c1[r] * scale + ((k1 >= q) ? slope_r : slope_l) * d1;
      // online softmax (reduce across the 16 lanes holding this row)
      float hm = fmaxf(s0, s1);
#pragma unroll
      for (int off = 1; off < 16; off <<= 1) hm = fmaxf(hm, __shfl_xor(hm, off, 16));
      float mn = fmaxf(m[r], hm);
      float corr = __expf(m[r] - mn);
      float p0 = __expf(s0 - mn), p1 = __expf(s1 - mn);
      float ps = p0 + p1;
#pragma unroll
      for (int off = 1; off < 16; off <<= 1) ps += __shfl_xor(ps, off, 16);
      l[r] = l[r] * corr + ps;
      m[r] = mn;
      o0[r] *= corr; o1[r] *= corr;
      int row = r + 8 * half;
      P[row * 32 + col] = (_Float16)p0;
      P[row * 32 + 16 + col] = (_Float16)p1;
    }
    __syncthreads();

    // --- P(16x32) @ V(32x32): A from LDS, B = V^T rows (contiguous) -------
    v16h pa = load_tile(P, 32, lane);
    v16h vb0 = load_tile(xnT + vb_base + kbase, SEQ, lane);
    v16h vb1 = load_tile(xnT + vb_base + (size_t)16 * SEQ + kbase, SEQ, lane);
    o0 = wmma16(pa, vb0, o0);
    o1 = wmma16(pa, vb1, o1);
    __syncthreads();
  }

  // normalize + residual add into x (each (row,dim) owned by exactly one lane)
#pragma unroll
  for (int r = 0; r < 8; ++r) {
    float inv = 1.0f / l[r];
    size_t base = bb + (size_t)(qbase + r + 8 * half) * EMB_DIM + hb;
    x[base + col]      += o0[r] * inv;
    x[base + 16 + col] += o1[r] * inv;
  }
}

// ---------------------------------------------------------------------------
// MLP GEMM 1: h = GELU(xn @ W1 + b1). A tile TDM-staged in LDS, shared by the
// block's 4 waves (16x64 output per block). W1 pre-transposed [MLP_H][EMB_DIM].
// ---------------------------------------------------------------------------
__global__ void mlp1_kernel(const _Float16* __restrict__ xn,
                            const _Float16* __restrict__ w1T,
                            const float* __restrict__ b1,
                            _Float16* __restrict__ h) {
  __shared__ _Float16 Atile[16 * 32];
  int lane = threadIdx.x & 31, wave = threadIdx.x >> 5;
  int row0 = blockIdx.x * 16;
  int col0 = (blockIdx.y * 4 + wave) * 16;
  const _Float16* wptr = w1T + (size_t)col0 * EMB_DIM;
  v8f acc = {};
#pragma unroll
  for (int kk = 0; kk < EMB_DIM; kk += 32) {
    if (wave == 0)
      tdm_load_tile16x32(Atile, xn + (size_t)row0 * EMB_DIM + kk, EMB_DIM, lane);
    if (kk + 32 < EMB_DIM) __builtin_prefetch(wptr + kk + 32, 0, 1);
    __syncthreads();
    v16h a = load_tile(Atile, 32, lane);
    v16h bt = load_tile(wptr + kk, EMB_DIM, lane);
    acc = wmma16(a, bt, acc);
    __syncthreads();
  }
  int half = lane >> 4, col = col0 + (lane & 15);
  float bias = b1[col];
#pragma unroll
  for (int r = 0; r < 8; ++r) {
    float v = acc[r] + bias;
    float g = 0.5f * v * (1.0f + erff(v * 0.70710678118654752f));  // exact GELU
    h[(size_t)(row0 + r + 8 * half) * MLP_H + col] = (_Float16)g;
  }
}

// ---------------------------------------------------------------------------
// MLP GEMM 2: x += h @ W2 + b2. Same structure; W2 pre-transposed [EMB_DIM][MLP_H].
// ---------------------------------------------------------------------------
__global__ void mlp2_kernel(const _Float16* __restrict__ h,
                            const _Float16* __restrict__ w2T,
                            const float* __restrict__ b2,
                            float* __restrict__ x) {
  __shared__ _Float16 Atile[16 * 32];
  int lane = threadIdx.x & 31, wave = threadIdx.x >> 5;
  int row0 = blockIdx.x * 16;
  int col0 = (blockIdx.y * 4 + wave) * 16;
  const _Float16* wptr = w2T + (size_t)col0 * MLP_H;
  v8f acc = {};
  for (int kk = 0; kk < MLP_H; kk += 32) {
    if (wave == 0)
      tdm_load_tile16x32(Atile, h + (size_t)row0 * MLP_H + kk, MLP_H, lane);
    if (kk + 32 < MLP_H) __builtin_prefetch(wptr + kk + 32, 0, 1);
    __syncthreads();
    v16h a = load_tile(Atile, 32, lane);
    v16h bt = load_tile(wptr + kk, MLP_H, lane);
    acc = wmma16(a, bt, acc);
    __syncthreads();
  }
  int half = lane >> 4, col = col0 + (lane & 15);
  float bias = b2[col];
#pragma unroll
  for (int r = 0; r < 8; ++r)
    x[(size_t)(row0 + r + 8 * half) * EMB_DIM + col] += acc[r] + bias;
}

// ---------------------------------------------------------------------------
// Weight pre-transpose + f32->f16 (all layers at once)
// ---------------------------------------------------------------------------
__global__ void tr_w1_kernel(const float* __restrict__ w1, _Float16* __restrict__ w1T) {
  int idx = blockIdx.x * blockDim.x + threadIdx.x;
  if (idx >= N_LAYERS * EMB_DIM * MLP_H) return;
  int l = idx / (EMB_DIM * MLP_H);
  int rem = idx - l * (EMB_DIM * MLP_H);
  int d = rem / MLP_H;
  int c = rem - d * MLP_H;
  w1T[(size_t)l * EMB_DIM * MLP_H + (size_t)c * EMB_DIM + d] = (_Float16)w1[idx];
}

__global__ void tr_w2_kernel(const float* __restrict__ w2, _Float16* __restrict__ w2T) {
  int idx = blockIdx.x * blockDim.x + threadIdx.x;
  if (idx >= N_LAYERS * EMB_DIM * MLP_H) return;
  int l = idx / (EMB_DIM * MLP_H);
  int rem = idx - l * (EMB_DIM * MLP_H);
  int hh = rem / EMB_DIM;
  int d = rem - hh * EMB_DIM;
  w2T[(size_t)l * EMB_DIM * MLP_H + (size_t)d * MLP_H + hh] = (_Float16)w2[idx];
}

// ---------------------------------------------------------------------------
// Launcher
// ---------------------------------------------------------------------------
extern "C" void kernel_launch(void* const* d_in, const int* in_sizes, int n_in,
                              void* d_out, int out_size, void* d_ws, size_t ws_size,
                              hipStream_t stream) {
  (void)in_sizes; (void)n_in; (void)out_size; (void)ws_size;
  const int*   tokens  = (const int*)d_in[0];
  const int*   pos_ids = (const int*)d_in[1];
  const float* emb     = (const float*)d_in[2];
  const float* iw      = (const float*)d_in[3];
  const float* pw      = (const float*)d_in[4];
  const float* ln1_w   = (const float*)d_in[5];
  const float* ln1_b   = (const float*)d_in[6];
  const float* ln2_w   = (const float*)d_in[7];
  const float* ln2_b   = (const float*)d_in[8];
  const float* w1      = (const float*)d_in[9];
  const float* b1      = (const float*)d_in[10];
  const float* w2      = (const float*)d_in[11];
  const float* b2      = (const float*)d_in[12];
  const float* lnf_w   = (const float*)d_in[13];
  const float* lnf_b   = (const float*)d_in[14];

  char* ws = (char*)d_ws;
  float*    x   = (float*)ws;     ws += (size_t)ROWS * EMB_DIM * sizeof(float);
  _Float16* xn  = (_Float16*)ws;  ws += (size_t)ROWS * EMB_DIM * sizeof(_Float16);
  _Float16* xnT = (_Float16*)ws;  ws += (size_t)ROWS * EMB_DIM * sizeof(_Float16);
  _Float16* h   = (_Float16*)ws;  ws += (size_t)ROWS * MLP_H * sizeof(_Float16);
  _Float16* w1T = (_Float16*)ws;  ws += (size_t)N_LAYERS * EMB_DIM * MLP_H * sizeof(_Float16);
  _Float16* w2T = (_Float16*)ws;

  embed_kernel<<<(ROWS * EMB_DIM) / 256, 256, 0, stream>>>(tokens, pos_ids, emb, iw, pw, x);
  tr_w1_kernel<<<(N_LAYERS * EMB_DIM * MLP_H) / 256, 256, 0, stream>>>(w1, w1T);
  tr_w2_kernel<<<(N_LAYERS * EMB_DIM * MLP_H) / 256, 256, 0, stream>>>(w2, w2T);

  for (int l = 0; l < N_LAYERS; ++l) {
    ln_f16_kernel<<<ROWS / 8, 256, 0, stream>>>(
        x, ln1_w + l * EMB_DIM, ln1_b + l * EMB_DIM, xn, xnT, 1);
    attn_kernel<<<dim3(SEQ / 16, HEADS, BATCH), 32, 0, stream>>>(xn, xnT, x);
    ln_f16_kernel<<<ROWS / 8, 256, 0, stream>>>(
        x, ln2_w + l * EMB_DIM, ln2_b + l * EMB_DIM, xn, xnT, 0);
    mlp1_kernel<<<dim3(ROWS / 16, MLP_H / 64), 128, 0, stream>>>(
        xn, w1T + (size_t)l * EMB_DIM * MLP_H, b1 + l * MLP_H, h);
    mlp2_kernel<<<dim3(ROWS / 16, EMB_DIM / 64), 128, 0, stream>>>(
        h, w2T + (size_t)l * EMB_DIM * MLP_H, b2 + l * EMB_DIM, x);
  }

  lnf_kernel<<<ROWS / 8, 256, 0, stream>>>(x, lnf_w, lnf_b, (float*)d_out);
}
